// EncoderLayer_79216376808250
// MI455X (gfx1250) — matrix operand order repaired and checked by reading, not compiled, not run on hardware
//
#include <hip/hip_runtime.h>
#include <hip/hip_bf16.h>
#include <stdint.h>

#define D_MODEL 512
#define N_HEADS 8
#define HEAD_DIM 64
#define D_FF 2048
#define BATCH 64
#define SEQ 500
#define TOKENS (BATCH * SEQ)   // 32000, multiple of 128

typedef __attribute__((ext_vector_type(16))) _Float16 v16h;
typedef __attribute__((ext_vector_type(8)))  _Float16 v8h;
typedef __attribute__((ext_vector_type(8)))  float    v8f;

// ---- WMMA helpers (CDNA5 16x16x32 f16 -> f32) --------------------------------
static __device__ inline v8f wmma_f16(v16h a, v16h b, v8f c) {
  // (neg_a, A, neg_b, B, c_mod, C, reuse_a, reuse_b)
  return __builtin_amdgcn_wmma_f32_16x16x32_f16(false, a, false, b, (short)0, c,
                                                false, false);
}

// Build a 16x32 A-fragment from an LDS row (row-major, this lane's row).
// Contiguous K: lanes 0-15 hold K {0..15}; lanes 16-31 hold K {16..31}.
static __device__ inline v16h make_a_frag(const _Float16* rowbase, int hi) {
  v8h lo = *(const v8h*)(rowbase + 16 * hi);
  v8h hh = *(const v8h*)(rowbase + 8 + 16 * hi);
  return __builtin_shufflevector(lo, hh, 0, 1, 2, 3, 4, 5, 6, 7,
                                 8, 9, 10, 11, 12, 13, 14, 15);
}

// CDNA5 async copy: 16 bytes global -> LDS, tracked by ASYNCcnt.
static __device__ inline void async_copy_16B(const _Float16* gsrc, _Float16* ldst) {
  unsigned lds_off = (unsigned)(uintptr_t)ldst;        // low 32 bits = LDS offset
  unsigned long long ga = (unsigned long long)(uintptr_t)gsrc;
  asm volatile("global_load_async_to_lds_b128 %0, %1, off"
               :: "v"(lds_off), "v"(ga) : "memory");
}
static __device__ inline void wait_async_all() {
  asm volatile("s_wait_asynccnt 0x0" ::: "memory");
}

// ---- VALU cross-lane reductions within 16-lane rows (v_permlane16_b32) -------
// xor-butterfly selectors: per-lane source nibble = lane ^ k, packed LSB-first.
#if __has_builtin(__builtin_amdgcn_permlane16)
static __device__ inline float xor16f(float v, unsigned lo, unsigned hi2) {
  unsigned u = __float_as_uint(v);
  unsigned r = __builtin_amdgcn_permlane16(u, u, lo, hi2, false, false);
  return __uint_as_float(r);
}
static __device__ inline float rowmax16(float v) {
  v = fmaxf(v, xor16f(v, 0x67452301u, 0xEFCDAB89u));  // xor 1
  v = fmaxf(v, xor16f(v, 0x54761032u, 0xDCFE98BAu));  // xor 2
  v = fmaxf(v, xor16f(v, 0x32107654u, 0xBA98FEDCu));  // xor 4
  v = fmaxf(v, xor16f(v, 0xFEDCBA98u, 0x76543210u));  // xor 8
  return v;
}
static __device__ inline float rowsum16(float v) {
  v += xor16f(v, 0x67452301u, 0xEFCDAB89u);
  v += xor16f(v, 0x54761032u, 0xDCFE98BAu);
  v += xor16f(v, 0x32107654u, 0xBA98FEDCu);
  v += xor16f(v, 0xFEDCBA98u, 0x76543210u);
  return v;
}
#else
static __device__ inline float rowmax16(float v) {
#pragma unroll
  for (int o = 8; o > 0; o >>= 1) v = fmaxf(v, __shfl_xor(v, o, 16));
  return v;
}
static __device__ inline float rowsum16(float v) {
#pragma unroll
  for (int o = 8; o > 0; o >>= 1) v += __shfl_xor(v, o, 16);
  return v;
}
#endif

// ---- fp32 -> fp16 conversion -------------------------------------------------
__global__ void f32_to_f16_kernel(const float* __restrict__ src,
                                  _Float16* __restrict__ dst, int n) {
  int i = blockIdx.x * blockDim.x + threadIdx.x;
  if (i < n) dst[i] = (_Float16)src[i];
}

// ---- LayerNorm (optionally fused residual add), emits f32 + f16 --------------
__global__ __launch_bounds__(128) void ln_kernel(const float* __restrict__ in0,
                                                 const float* __restrict__ in1,
                                                 const float* __restrict__ g,
                                                 const float* __restrict__ beta,
                                                 float* __restrict__ out_f32,
                                                 _Float16* __restrict__ out_f16) {
  __shared__ float sbuf[8];
  const int t = blockIdx.x;
  const int tid = threadIdx.x;
  const size_t base = (size_t)t * D_MODEL;
  float v[4];
  float s = 0.f, s2 = 0.f;
#pragma unroll
  for (int k = 0; k < 4; ++k) {
    int c = tid + k * 128;
    float xv = in0[base + c];
    if (in1) xv += in1[base + c];
    v[k] = xv; s += xv; s2 += xv * xv;
  }
#pragma unroll
  for (int o = 16; o > 0; o >>= 1) {
    s  += __shfl_xor(s, o, 32);
    s2 += __shfl_xor(s2, o, 32);
  }
  int wv = tid >> 5;
  if ((tid & 31) == 0) { sbuf[wv] = s; sbuf[4 + wv] = s2; }
  __syncthreads();
  s  = sbuf[0] + sbuf[1] + sbuf[2] + sbuf[3];
  s2 = sbuf[4] + sbuf[5] + sbuf[6] + sbuf[7];
  const float mu  = s * (1.f / D_MODEL);
  const float var = s2 * (1.f / D_MODEL) - mu * mu;
  const float r   = rsqrtf(var + 1e-6f);
#pragma unroll
  for (int k = 0; k < 4; ++k) {
    int c = tid + k * 128;
    float y = (v[k] - mu) * r * g[c] + beta[c];
    out_f32[base + c] = y;
    out_f16[base + c] = (_Float16)y;
  }
}

// ---- Causal flash attention (q=k=v=h), per (batch*head, 64-query block) ------
__global__ __launch_bounds__(128) void attn_kernel(const _Float16* __restrict__ h16,
                                                   float* __restrict__ out) {
  __shared__ _Float16 lds_q[64 * 64];     // queries, row-major [qrow][dim]
  __shared__ _Float16 lds_k[32 * 64];     // keys,    row-major [krow][dim]
  __shared__ _Float16 lds_vT[64 * 32];    // values,  transposed [dim][krow]
  __shared__ _Float16 lds_p[4 * 16 * 32]; // per-wave P staging (C->A relayout)

  const int b    = blockIdx.x / N_HEADS;
  const int hd   = blockIdx.x % N_HEADS;
  const int hOff = hd * HEAD_DIM;
  const int qblk = blockIdx.y;
  const int tid  = threadIdx.x;
  const int w    = tid >> 5;
  const int n15  = tid & 15;
  const int hi   = (tid >> 4) & 1;

  const size_t tokBase = (size_t)b * SEQ;
  const int qTile0 = qblk * 64;

  // stage 64x64 Q tile with 16B vector loads (zero-pad rows >= SEQ)
#pragma unroll
  for (int i = 0; i < 4; ++i) {
    int chunk = tid + i * 128;          // 512 chunks of 8 halves
    int r = chunk >> 3, c0 = (chunk & 7) * 8;
    int qr = qTile0 + r;
    v8h val = {};
    if (qr < SEQ) val = *(const v8h*)(h16 + (tokBase + qr) * D_MODEL + hOff + c0);
    *(v8h*)(lds_q + r * 64 + c0) = val;
  }
  __syncthreads();

  const int q0 = qTile0 + w * 16;     // this wave's first query row
  v16h qa0 = make_a_frag(lds_q + (w * 16 + n15) * 64 + 0,  hi);
  v16h qa1 = make_a_frag(lds_q + (w * 16 + n15) * 64 + 32, hi);

  v8f o0 = {}, o1 = {}, o2 = {}, o3 = {};
  float mrun[8], lrun[8];
#pragma unroll
  for (int m = 0; m < 8; ++m) { mrun[m] = -1e30f; lrun[m] = 0.f; }

  _Float16* pW = lds_p + w * 16 * 32;

  int kEnd = qTile0 + 64;
  if (kEnd > SEQ) kEnd = SEQ;

  for (int j = 0; j < kEnd; j += 32) {
    __syncthreads();   // previous iteration's consumers done before restaging
#pragma unroll
    for (int i = 0; i < 2; ++i) {
      int chunk = tid + i * 128;        // 256 chunks of 8 halves
      int r = chunk >> 3, c0 = (chunk & 7) * 8;
      int kr = j + r;
      v8h val = {};
      if (kr < SEQ) val = *(const v8h*)(h16 + (tokBase + kr) * D_MODEL + hOff + c0);
      *(v8h*)(lds_k + r * 64 + c0) = val;   // row-major for QK^T B-fragments
#pragma unroll
      for (int e = 0; e < 8; ++e) lds_vT[(c0 + e) * 32 + r] = val[e];
    }
    __syncthreads();
    if (j > q0 + 15) continue;   // wave-uniform: everything masked for this wave

    // scores S = Q * K^T : two 16x16 tiles (keys j..j+15 and j+16..j+31)
    v8f s0 = {}, s1 = {};
    v16h kb;
    kb = *(const v16h*)(lds_k + n15 * 64 + 16 * hi);
    s0 = wmma_f16(qa0, kb, s0);
    kb = *(const v16h*)(lds_k + n15 * 64 + 32 + 16 * hi);
    s0 = wmma_f16(qa1, kb, s0);
    kb = *(const v16h*)(lds_k + (n15 + 16) * 64 + 16 * hi);
    s1 = wmma_f16(qa0, kb, s1);
    kb = *(const v16h*)(lds_k + (n15 + 16) * 64 + 32 + 16 * hi);
    s1 = wmma_f16(qa1, kb, s1);

    const float scale = 0.125f;   // 1/sqrt(64)
#pragma unroll
    for (int m = 0; m < 8; ++m) {
      const int qr = q0 + m + 8 * hi;
      float sa = s0[m] * scale;
      float sb = s1[m] * scale;
      if (j + n15 > qr      || qr >= SEQ) sa = -1e30f;
      if (j + 16 + n15 > qr || qr >= SEQ) sb = -1e30f;
      const float tm = rowmax16(fmaxf(sa, sb));
      const float nm = fmaxf(mrun[m], tm);
      const float cf = __expf(mrun[m] - nm);
      const float p0 = __expf(sa - nm);
      const float p1 = __expf(sb - nm);
      const float rs = rowsum16(p0 + p1);
      lrun[m] = lrun[m] * cf + rs;
      mrun[m] = nm;
      o0[m] *= cf; o1[m] *= cf; o2[m] *= cf; o3[m] *= cf;
      pW[(m + 8 * hi) * 32 + n15]      = (_Float16)p0;
      pW[(m + 8 * hi) * 32 + 16 + n15] = (_Float16)p1;
    }
    // per-wave LDS round trip: C-layout P -> A-layout fragment
    asm volatile("s_wait_dscnt 0x0" ::: "memory");
    v16h pa = make_a_frag(pW + n15 * 32, hi);
    v16h vb;
    vb = *(const v16h*)(lds_vT + (0 * 16 + n15) * 32 + 16 * hi);
    o0 = wmma_f16(pa, vb, o0);
    vb = *(const v16h*)(lds_vT + (1 * 16 + n15) * 32 + 16 * hi);
    o1 = wmma_f16(pa, vb, o1);
    vb = *(const v16h*)(lds_vT + (2 * 16 + n15) * 32 + 16 * hi);
    o2 = wmma_f16(pa, vb, o2);
    vb = *(const v16h*)(lds_vT + (3 * 16 + n15) * 32 + 16 * hi);
    o3 = wmma_f16(pa, vb, o3);
  }

#pragma unroll
  for (int m = 0; m < 8; ++m) {
    const int qr = q0 + m + 8 * hi;
    if (qr >= SEQ) continue;
    const float inv = 1.f / lrun[m];
    const size_t rowp = (tokBase + qr) * D_MODEL + hOff;
    out[rowp + 0 * 16 + n15] = o0[m] * inv;
    out[rowp + 1 * 16 + n15] = o1[m] * inv;
    out[rowp + 2 * 16 + n15] = o2[m] * inv;
    out[rowp + 3 * 16 + n15] = o3[m] * inv;
  }
}

// ---- Double-buffered WMMA GEMM with async-to-LDS A staging -------------------
// C[M,N] = A[M,K] @ W[K,N] (+bias, relu -> f16 out | +resid -> f32 out)
template <int KDIM, int NDIM, bool RELU>
__global__ __launch_bounds__(256) void gemm_kernel(const _Float16* __restrict__ A,
                                                   const _Float16* __restrict__ W,
                                                   const float* __restrict__ bias,
                                                   const float* __restrict__ resid,
                                                   _Float16* __restrict__ outH,
                                                   float* __restrict__ outF) {
  __shared__ _Float16 lds_a[2][128 * 32];   // [row][k]  row-major
  __shared__ _Float16 lds_bT[2][128 * 32];  // [col][k]  W tile transposed

  const int tid = threadIdx.x;
  const int n15 = tid & 15;
  const int hi  = (tid >> 4) & 1;
  const int wv  = tid >> 5;       // 8 waves: 4 (M) x 2 (N)
  const int wm  = wv >> 1;
  const int wn  = wv & 1;
  const int rowBase = blockIdx.x * 128;
  const int colBase = blockIdx.y * 128;

  v8f zero = {};
  v8f acc[2][4];
#pragma unroll
  for (int i = 0; i < 2; ++i)
#pragma unroll
    for (int j = 0; j < 4; ++j) acc[i][j] = zero;

  // A tile: 128x32 halves = 512 x 16B chunks, 2 async copies per thread
  auto stageA = [&](int buf, int kk) {
#pragma unroll
    for (int i = 0; i < 2; ++i) {
      int chunk = tid + i * 256;
      int r  = chunk >> 2;
      int c0 = (chunk & 3) * 8;
      async_copy_16B(A + (size_t)(rowBase + r) * KDIM + kk + c0,
                     &lds_a[buf][r * 32 + c0]);
    }
  };
  // W tile: 32x128, transposed into LDS; coalesced v8h global reads
  auto stageB = [&](int buf, int kk) {
#pragma unroll
    for (int i = 0; i < 2; ++i) {
      int chunk = tid + i * 256;
      int k  = chunk >> 4;
      int c0 = (chunk & 15) * 8;
      v8h t = *(const v8h*)(W + (size_t)(kk + k) * NDIM + colBase + c0);
#pragma unroll
      for (int e = 0; e < 8; ++e) lds_bT[buf][(c0 + e) * 32 + k] = t[e];
    }
  };

  stageA(0, 0);
  stageB(0, 0);
  wait_async_all();
  __syncthreads();

  int buf = 0;
  for (int kk = 0; kk < KDIM; kk += 32) {
    const int nxt = buf ^ 1;
    if (kk + 32 < KDIM) {   // prefetch next tile into the other LDS buffer
      stageA(nxt, kk + 32);
      stageB(nxt, kk + 32);
    }
    if (kk + 64 < KDIM) {   // L2 prefetch two tiles ahead (global_prefetch_b8)
      __builtin_prefetch(A + (size_t)(rowBase + (tid >> 1)) * KDIM + kk + 64, 0, 1);
      __builtin_prefetch(W + (size_t)(kk + 64 + (tid >> 4)) * NDIM + colBase +
                             (tid & 15) * 8, 0, 1);
    }

    v16h af0 = make_a_frag(&lds_a[buf][(wm * 32 + 0  + n15) * 32], hi);
    v16h af1 = make_a_frag(&lds_a[buf][(wm * 32 + 16 + n15) * 32], hi);
#pragma unroll
    for (int ni = 0; ni < 4; ++ni) {
      v16h bf = *(const v16h*)(&lds_bT[buf][(wn * 64 + ni * 16 + n15) * 32 + 16 * hi]);
      acc[0][ni] = wmma_f16(af0, bf, acc[0][ni]);
      acc[1][ni] = wmma_f16(af1, bf, acc[1][ni]);
    }

    wait_async_all();     // next-buffer async copies landed (no-op on last iter)
    __syncthreads();      // all waves done reading current + staging next
    buf = nxt;
  }

#pragma unroll
  for (int mi = 0; mi < 2; ++mi)
#pragma unroll
    for (int ni = 0; ni < 4; ++ni)
#pragma unroll
      for (int m = 0; m < 8; ++m) {
        int r = rowBase + wm * 32 + mi * 16 + m + 8 * hi;
        int c = colBase + wn * 64 + ni * 16 + n15;
        float v = acc[mi][ni][m] + bias[c];
        if (RELU) {
          v = fmaxf(v, 0.f);
          outH[(size_t)r * NDIM + c] = (_Float16)v;
        } else {
          outF[(size_t)r * NDIM + c] = v + resid[(size_t)r * NDIM + c];
        }
      }
}

// ------------------------------------------------------------------------------
extern "C" void kernel_launch(void* const* d_in, const int* in_sizes, int n_in,
                              void* d_out, int out_size, void* d_ws, size_t ws_size,
                              hipStream_t stream) {
  (void)in_sizes; (void)n_in; (void)out_size; (void)ws_size;
  const float* x    = (const float*)d_in[0];
  const float* ln1g = (const float*)d_in[1];
  const float* ln1b = (const float*)d_in[2];
  const float* ln2g = (const float*)d_in[3];
  const float* ln2b = (const float*)d_in[4];
  const float* W1   = (const float*)d_in[5];
  const float* b1   = (const float*)d_in[6];
  const float* W2   = (const float*)d_in[7];
  const float* b2   = (const float*)d_in[8];
  float* out = (float*)d_out;

  char* ws = (char*)d_ws;
  size_t off = 0;
  auto carve = [&](size_t bytes) -> char* {
    char* p = ws + off;
    off += (bytes + 255) & ~(size_t)255;
    return p;
  };
  float*    hln_f32  = (float*)   carve((size_t)TOKENS * D_MODEL * 4);
  float*    attn_f32 = (float*)   carve((size_t)TOKENS * D_MODEL * 4);
  float*    h2_f32   = (float*)   carve((size_t)TOKENS * D_MODEL * 4);
  _Float16* hln_f16  = (_Float16*)carve((size_t)TOKENS * D_MODEL * 2);
  _Float16* h2_f16   = (_Float16*)carve((size_t)TOKENS * D_MODEL * 2);
  _Float16* act_f16  = (_Float16*)carve((size_t)TOKENS * D_FF * 2);
  _Float16* w1h      = (_Float16*)carve((size_t)D_MODEL * D_FF * 2);
  _Float16* w2h      = (_Float16*)carve((size_t)D_FF * D_MODEL * 2);

  const int nw = D_MODEL * D_FF;
  hipLaunchKernelGGL(f32_to_f16_kernel, dim3((nw + 255) / 256), dim3(256), 0, stream,
                     W1, w1h, nw);
  hipLaunchKernelGGL(f32_to_f16_kernel, dim3((nw + 255) / 256), dim3(256), 0, stream,
                     W2, w2h, nw);
  // h = LN1(x)
  hipLaunchKernelGGL(ln_kernel, dim3(TOKENS), dim3(128), 0, stream,
                     x, (const float*)nullptr, ln1g, ln1b, hln_f32, hln_f16);
  // attn(h)
  hipLaunchKernelGGL(attn_kernel, dim3(BATCH * N_HEADS, 8), dim3(128), 0, stream,
                     hln_f16, attn_f32);
  // h2 = LN2(h + attn)
  hipLaunchKernelGGL(ln_kernel, dim3(TOKENS), dim3(128), 0, stream,
                     hln_f32, attn_f32, ln2g, ln2b, h2_f32, h2_f16);
  // act = relu(h2 @ W1 + b1)
  hipLaunchKernelGGL((gemm_kernel<D_MODEL, D_FF, true>),
                     dim3(TOKENS / 128, D_FF / 128), dim3(256), 0, stream,
                     h2_f16, w1h, b1, (const float*)nullptr, act_f16, (float*)nullptr);
  // out = h2 + act @ W2 + b2
  hipLaunchKernelGGL((gemm_kernel<D_FF, D_MODEL, false>),
                     dim3(TOKENS / 128, D_MODEL / 128), dim3(256), 0, stream,
                     act_f16, w2h, b2, h2_f32, (_Float16*)nullptr, out);
}
